// MultiHeadAttention_12180527251811
// MI455X (gfx1250) — matrix operand order repaired and checked by reading, not compiled, not run on hardware
//
#include <hip/hip_runtime.h>

// ---------------------------------------------------------------------------
// MI455X (gfx1250, wave32) fused multi-head attention:
//   1) cvt:  fp32 weights -> f16 (native layout already B^T[N][K])
//   2) gemm_f32a_f16c: Q/K/V projections (fp32 activations -> f16 on the fly,
//      f32 WMMA accumulate, f16 outputs; softmax scale folded into Q)
//   3) attn_kernel: flash-attention (streaming softmax, causal+padding mask,
//      WMMA for Q*K^T and P*V, no materialized SxS scores)
//   4) gemm_f16a_f32c: output projection -> fp32 d_out
// f16 tile copies with no data transformation go through
// GLOBAL_LOAD_ASYNC_TO_LDS_B128 (ASYNCcnt) instead of VGPR staging.
// ---------------------------------------------------------------------------

typedef __attribute__((ext_vector_type(4)))  _Float16 v4h;
typedef __attribute__((ext_vector_type(8)))  _Float16 v8h;
typedef __attribute__((ext_vector_type(16))) _Float16 v16h;
typedef __attribute__((ext_vector_type(8)))  float    v8f;

union V16U { v16h v; v8h h[2]; };

#define NEGV (-1.0e30f)

// ds_swizzle xor within 32 lanes: offset = {0, xor[14:10], or[9:5]=0, and[4:0]=0x1f}
#define SWZ_XOR_F(x, m) \
  __int_as_float(__builtin_amdgcn_ds_swizzle(__float_as_int(x), (((m) << 10) | 0x1f)))

__device__ __forceinline__ v8f wmma16x16x32(v16h a, v16h b, v8f c) {
  // D = A(16x32,f16) * B(32x16,f16) + C(16x16,f32)
  return __builtin_amdgcn_wmma_f32_16x16x32_f16(false, a, false, b, (short)0, c,
                                                false, false);
}

// Async DMA: 16B global -> LDS, tracked by ASYNCcnt (no VGPR staging).
// Generic pointers to LDS carry the LDS byte offset in their low 32 bits.
__device__ __forceinline__ void async_copy_b128(_Float16* lds_dst,
                                                const _Float16* gsrc) {
  const unsigned ldsa = (unsigned)(unsigned long long)lds_dst;
  asm volatile("global_load_async_to_lds_b128 %0, %1, off"
               :
               : "v"(ldsa), "v"((unsigned long long)gsrc)
               : "memory");
}
__device__ __forceinline__ void wait_async0() {
  asm volatile("s_wait_asynccnt 0" ::: "memory");
}

// A-fragment (16x32 f16) from row-major tile [M][ld]:
//  lanes 0-15: row = m0+lane,  K = 0..7 and 16..23
//  lanes16-31: row = m0+lane-16, K = 8..15 and 24..31
__device__ __forceinline__ v16h load_a_frag(const _Float16* tile, int m0, int kk0,
                                            int ld) {
  const int lane = threadIdx.x & 31;
  const _Float16* p =
      tile + (size_t)(m0 + (lane & 15)) * ld + kk0 + ((lane >> 4) << 3);
  V16U u;
  u.h[0] = *(const v8h*)(p);
  u.h[1] = *(const v8h*)(p + 16);
  return u.v;
}

// B-fragment (32x16 f16) from transposed storage B^T[N][ld] (K contiguous):
//  lanes 0-15: col = n0+lane,  K = kk0+0..7 and kk0+16..23
//  lanes16-31: col = n0+lane-16, K = kk0+8..15 and kk0+24..31
__device__ __forceinline__ v16h load_b_frag(const _Float16* tile, int n0, int kk0,
                                            int ld) {
  const int lane = threadIdx.x & 31;
  const _Float16* p =
      tile + (size_t)(n0 + (lane & 15)) * ld + kk0 + ((lane >> 4) << 3);
  V16U u;
  u.h[0] = *(const v8h*)(p);
  u.h[1] = *(const v8h*)(p + 16);
  return u.v;
}

// ---------------------------------------------------------------------------
// Weight conversion: four 1M-element fp32 tensors -> f16 (layout preserved).
// ---------------------------------------------------------------------------
__global__ __launch_bounds__(256) void cvt_f32_f16(
    const float* __restrict__ s0, const float* __restrict__ s1,
    const float* __restrict__ s2, const float* __restrict__ s3,
    _Float16* __restrict__ d0, _Float16* __restrict__ d1,
    _Float16* __restrict__ d2, _Float16* __restrict__ d3, int n) {
  const float* s;
  _Float16* d;
  switch (blockIdx.y) {
    case 0:  s = s0; d = d0; break;
    case 1:  s = s1; d = d1; break;
    case 2:  s = s2; d = d2; break;
    default: s = s3; d = d3; break;
  }
  const int i = (blockIdx.x * 256 + threadIdx.x) << 2;
  if (i < n) {
    float4 f = *(const float4*)(s + i);
    v4h hv = {(_Float16)f.x, (_Float16)f.y, (_Float16)f.z, (_Float16)f.w};
    *(v4h*)(d + i) = hv;
  }
}

// ---------------------------------------------------------------------------
// GEMM: C[M,N](f16) = ascale*A[M,K](f32) * B^T[N,K](f16)
// Block tile 128x128, BK=32, 256 threads = 8 waves (4 over M x 2 over N),
// each wave owns a 32x64 sub-tile (2x4 WMMA accumulators).
// B panel copied with async DMA; A converted fp32->f16 through VGPRs.
// ---------------------------------------------------------------------------
#define BM 128
#define BN 128
#define BKK 32

__global__ __launch_bounds__(256) void gemm_f32a_f16c(
    const float* __restrict__ A, const _Float16* __restrict__ Bt,
    _Float16* __restrict__ C, int M, int N, int K, float ascale) {
  __shared__ _Float16 As[BM * BKK];
  __shared__ _Float16 Bs[BN * BKK];
  const int tid = threadIdx.x;
  const int lane = tid & 31;
  const int wave = tid >> 5;
  const int m0 = blockIdx.y * BM;
  const int n0 = blockIdx.x * BN;
  const int wm = (wave & 3) << 5;  // 0,32,64,96
  const int wn = (wave >> 2) << 6; // 0,64
  v8f acc[2][4] = {};
  const int arow = tid >> 1;        // 0..127
  const int acb = (tid & 1) << 4;   // 0 or 16

  for (int k0 = 0; k0 < K; k0 += BKK) {
    // B tile: 128 cols x 32 K halves, K-contiguous -> async DMA to LDS
    const _Float16* gb = Bt + (size_t)(n0 + arow) * K + k0 + acb;
    async_copy_b128(&Bs[arow * BKK + acb], gb);
    async_copy_b128(&Bs[arow * BKK + acb + 8], gb + 8);
    // A tile: 128x32 fp32 -> f16 (scaled) into LDS
    const float* ga = A + (size_t)(m0 + arow) * K + k0 + acb;
#pragma unroll
    for (int i = 0; i < 4; ++i) {
      float4 f = *(const float4*)(ga + (i << 2));
      v4h hv = {(_Float16)(f.x * ascale), (_Float16)(f.y * ascale),
                (_Float16)(f.z * ascale), (_Float16)(f.w * ascale)};
      *(v4h*)&As[arow * BKK + acb + (i << 2)] = hv;
    }
    if (k0 + BKK < K) {
      __builtin_prefetch(ga + BKK, 0, 1);
      __builtin_prefetch(gb + BKK, 0, 1);
    }
    wait_async0();
    __syncthreads();

    v16h a0 = load_a_frag(As, wm, 0, BKK);
    v16h a1 = load_a_frag(As, wm + 16, 0, BKK);
    v16h bfr[4];
#pragma unroll
    for (int nt = 0; nt < 4; ++nt)
      bfr[nt] = load_b_frag(Bs, wn + (nt << 4), 0, BKK);
#pragma unroll
    for (int nt = 0; nt < 4; ++nt) {
      acc[0][nt] = wmma16x16x32(a0, bfr[nt], acc[0][nt]);
      acc[1][nt] = wmma16x16x32(a1, bfr[nt], acc[1][nt]);
    }
    __syncthreads();
  }

  const int coln = lane & 15;
  const int rb = (lane >> 4) << 3;
#pragma unroll
  for (int mt = 0; mt < 2; ++mt)
#pragma unroll
    for (int nt = 0; nt < 4; ++nt)
#pragma unroll
      for (int r = 0; r < 8; ++r)
        C[(size_t)(m0 + wm + mt * 16 + rb + r) * N +
          (n0 + wn + nt * 16 + coln)] = (_Float16)acc[mt][nt][r];
}

// ---------------------------------------------------------------------------
// GEMM: C[M,N](f32) = A[M,K](f16) * B^T[N,K](f16)   (output projection)
// Both panels are plain f16 copies -> async DMA to LDS.
// ---------------------------------------------------------------------------
__global__ __launch_bounds__(256) void gemm_f16a_f32c(
    const _Float16* __restrict__ A, const _Float16* __restrict__ Bt,
    float* __restrict__ C, int M, int N, int K) {
  __shared__ _Float16 As[BM * BKK];
  __shared__ _Float16 Bs[BN * BKK];
  const int tid = threadIdx.x;
  const int lane = tid & 31;
  const int wave = tid >> 5;
  const int m0 = blockIdx.y * BM;
  const int n0 = blockIdx.x * BN;
  const int wm = (wave & 3) << 5;
  const int wn = (wave >> 2) << 6;
  v8f acc[2][4] = {};
  const int arow = tid >> 1;
  const int acb = (tid & 1) << 4;

  for (int k0 = 0; k0 < K; k0 += BKK) {
    const _Float16* ga = A + (size_t)(m0 + arow) * K + k0 + acb;
    async_copy_b128(&As[arow * BKK + acb], ga);
    async_copy_b128(&As[arow * BKK + acb + 8], ga + 8);
    const _Float16* gb = Bt + (size_t)(n0 + arow) * K + k0 + acb;
    async_copy_b128(&Bs[arow * BKK + acb], gb);
    async_copy_b128(&Bs[arow * BKK + acb + 8], gb + 8);
    if (k0 + BKK < K) {
      __builtin_prefetch(ga + BKK, 0, 1);
      __builtin_prefetch(gb + BKK, 0, 1);
    }
    wait_async0();
    __syncthreads();

    v16h a0 = load_a_frag(As, wm, 0, BKK);
    v16h a1 = load_a_frag(As, wm + 16, 0, BKK);
    v16h bfr[4];
#pragma unroll
    for (int nt = 0; nt < 4; ++nt)
      bfr[nt] = load_b_frag(Bs, wn + (nt << 4), 0, BKK);
#pragma unroll
    for (int nt = 0; nt < 4; ++nt) {
      acc[0][nt] = wmma16x16x32(a0, bfr[nt], acc[0][nt]);
      acc[1][nt] = wmma16x16x32(a1, bfr[nt], acc[1][nt]);
    }
    __syncthreads();
  }

  const int coln = lane & 15;
  const int rb = (lane >> 4) << 3;
#pragma unroll
  for (int mt = 0; mt < 2; ++mt)
#pragma unroll
    for (int nt = 0; nt < 4; ++nt)
#pragma unroll
      for (int r = 0; r < 8; ++r)
        C[(size_t)(m0 + wm + mt * 16 + rb + r) * N +
          (n0 + wn + nt * 16 + coln)] = acc[mt][nt][r];
}

// ---------------------------------------------------------------------------
// Flash attention: block = 128 query rows (8 waves x 16 rows), KV tiles of 64.
// Q/K/V are f16 [B*S][1024] with head h at columns h*64..h*64+63.
// Q was pre-scaled by 1/sqrt(64) in the projection.
// K tile is a straight copy -> async DMA; V tile is transposed element-wise
// through VGPRs. Row softmax stats are wave-local (C layout: vgpr r -> rows
// r / r+8 in the two 16-lane halves); reductions via ds_swizzle xor 1/2/4/8.
// ---------------------------------------------------------------------------
#define ABR 128
#define ABC 64

__global__ __launch_bounds__(256) void attn_kernel(
    const _Float16* __restrict__ Q, const _Float16* __restrict__ Kb,
    const _Float16* __restrict__ Vb, const unsigned char* __restrict__ pmask,
    _Float16* __restrict__ Ctx, int S) {
  __shared__ _Float16 Ks[ABC * 64];      // K tile  [t][k]  (B^T layout for QK^T)
  __shared__ _Float16 Vs[64 * ABC];      // V tile  [v][t]  (B^T layout for P*V)
  __shared__ _Float16 Ps[8 * 16 * 64];   // per-wave P scratch [row][t]

  const int tid = threadIdx.x, lane = tid & 31, wave = tid >> 5;
  const int b = blockIdx.z, h = blockIdx.y;
  const int m0 = blockIdx.x * ABR;
  const int mw = m0 + (wave << 4);       // this wave's first query row
  const int LD = 1024;                   // H*HK
  const _Float16* Qh = Q + (size_t)b * S * LD + h * 64;
  const _Float16* Kh = Kb + (size_t)b * S * LD + h * 64;
  const _Float16* Vh = Vb + (size_t)b * S * LD + h * 64;
  const unsigned char* pmb = pmask + (size_t)b * S * S;

  // Q fragments (16 rows x K=64 -> two 16x32 frags), loaded once from global
  v16h qf0, qf1;
  {
    const _Float16* p =
        Qh + (size_t)(mw + (lane & 15)) * LD + ((lane >> 4) << 3);
    V16U u;
    u.h[0] = *(const v8h*)(p);
    u.h[1] = *(const v8h*)(p + 16);
    qf0 = u.v;
    u.h[0] = *(const v8h*)(p + 32);
    u.h[1] = *(const v8h*)(p + 48);
    qf1 = u.v;
  }

  float mrun[8], lrun[8];
  v8f Oacc[4] = {};
#pragma unroll
  for (int r = 0; r < 8; ++r) {
    mrun[r] = -__builtin_inff();
    lrun[r] = 0.f;
  }

  const int coln = lane & 15;
  const int rb = (lane >> 4) << 3;
  _Float16* myP = Ps + wave * (16 * 64);

  int tend = m0 + ABR;                   // causal bound for this block
  if (tend > S) tend = S;

  for (int t0 = 0; t0 < tend; t0 += ABC) {
    // ---- K tile [t][k] via async DMA; V tile transposed [v][t] via VGPRs ----
    {
      const int tr = tid >> 2;           // t row 0..63
      const int cb = (tid & 3) << 4;     // 16-half chunk
      const _Float16* gk = Kh + (size_t)(t0 + tr) * LD + cb;
      async_copy_b128(&Ks[tr * 64 + cb], gk);
      async_copy_b128(&Ks[tr * 64 + cb + 8], gk + 8);
      const _Float16* gv = Vh + (size_t)(t0 + tr) * LD + cb;
      v8h x0 = *(const v8h*)gv;
      v8h x1 = *(const v8h*)(gv + 8);
#pragma unroll
      for (int i = 0; i < 8; ++i) Vs[(cb + i) * 64 + tr] = x0[i];
#pragma unroll
      for (int i = 0; i < 8; ++i) Vs[(cb + 8 + i) * 64 + tr] = x1[i];
      if (t0 + ABC < tend) {
        __builtin_prefetch(gk + (size_t)ABC * LD, 0, 1);
        __builtin_prefetch(gv + (size_t)ABC * LD, 0, 1);
      }
    }
    wait_async0();
    __syncthreads();

    // ---- scores S = Q * K^T (16x64 per wave, 4 n-tiles x 2 k-steps) ----
    v8f sacc[4] = {};
    {
      v16h bk[4];
#pragma unroll
      for (int nt = 0; nt < 4; ++nt) bk[nt] = load_b_frag(Ks, nt << 4, 0, 64);
#pragma unroll
      for (int nt = 0; nt < 4; ++nt) sacc[nt] = wmma16x16x32(qf0, bk[nt], sacc[nt]);
#pragma unroll
      for (int nt = 0; nt < 4; ++nt) bk[nt] = load_b_frag(Ks, nt << 4, 32, 64);
#pragma unroll
      for (int nt = 0; nt < 4; ++nt) sacc[nt] = wmma16x16x32(qf1, bk[nt], sacc[nt]);
    }

    // ---- causal + padding mask ----
#pragma unroll
    for (int r = 0; r < 8; ++r) {
      const int mrow = mw + rb + r;
      const unsigned char* pr = pmb + (size_t)mrow * S + t0 + coln;
#pragma unroll
      for (int nt = 0; nt < 4; ++nt) {
        const int tcol = t0 + (nt << 4) + coln;
        const bool dead = (tcol > mrow) || (pr[nt << 4] != 0);
        float s = sacc[nt][r];
        sacc[nt][r] = dead ? NEGV : s;
      }
    }

    // ---- running softmax stats (per-row, wave-local) ----
    float alpha[8];
#pragma unroll
    for (int r = 0; r < 8; ++r) {
      float mx = fmaxf(fmaxf(sacc[0][r], sacc[1][r]),
                       fmaxf(sacc[2][r], sacc[3][r]));
      mx = fmaxf(mx, SWZ_XOR_F(mx, 1));
      mx = fmaxf(mx, SWZ_XOR_F(mx, 2));
      mx = fmaxf(mx, SWZ_XOR_F(mx, 4));
      mx = fmaxf(mx, SWZ_XOR_F(mx, 8));
      const float newm = fmaxf(mrun[r], mx);
      alpha[r] = __expf(mrun[r] - newm);
      mrun[r] = newm;
    }

    // ---- P = exp(S - m): rowsum + spill to per-wave LDS (f16) ----
#pragma unroll
    for (int r = 0; r < 8; ++r) {
      float rs = 0.f;
#pragma unroll
      for (int nt = 0; nt < 4; ++nt) {
        const float p = __expf(sacc[nt][r] - mrun[r]);
        rs += p;
        myP[(rb + r) * 64 + (nt << 4) + coln] = (_Float16)p;
      }
      rs += SWZ_XOR_F(rs, 1);
      rs += SWZ_XOR_F(rs, 2);
      rs += SWZ_XOR_F(rs, 4);
      rs += SWZ_XOR_F(rs, 8);
      lrun[r] = lrun[r] * alpha[r] + rs;
    }

    // ---- rescale running O, then O += P * V ----
#pragma unroll
    for (int vt = 0; vt < 4; ++vt)
#pragma unroll
      for (int r = 0; r < 8; ++r) Oacc[vt][r] = Oacc[vt][r] * alpha[r];

    v16h pa0 = load_a_frag(myP, 0, 0, 64);   // wave-local, in-order DS
    v16h pa1 = load_a_frag(myP, 0, 32, 64);
    {
      v16h vb[4];
#pragma unroll
      for (int vt = 0; vt < 4; ++vt) vb[vt] = load_b_frag(Vs, vt << 4, 0, 64);
#pragma unroll
      for (int vt = 0; vt < 4; ++vt) Oacc[vt] = wmma16x16x32(pa0, vb[vt], Oacc[vt]);
#pragma unroll
      for (int vt = 0; vt < 4; ++vt) vb[vt] = load_b_frag(Vs, vt << 4, 32, 64);
#pragma unroll
      for (int vt = 0; vt < 4; ++vt) Oacc[vt] = wmma16x16x32(pa1, vb[vt], Oacc[vt]);
    }
    __syncthreads();
  }

  // ---- epilogue: normalize and write ctx (f16, heads concatenated) ----
#pragma unroll
  for (int vt = 0; vt < 4; ++vt)
#pragma unroll
    for (int r = 0; r < 8; ++r) {
      const float o = Oacc[vt][r] / lrun[r];
      Ctx[(size_t)b * S * LD + (size_t)(mw + rb + r) * LD + h * 64 +
          (vt << 4) + coln] = (_Float16)o;
    }
}

// ---------------------------------------------------------------------------
// Host launcher
// ---------------------------------------------------------------------------
extern "C" void kernel_launch(void* const* d_in, const int* in_sizes, int n_in,
                              void* d_out, int out_size, void* d_ws,
                              size_t ws_size, hipStream_t stream) {
  (void)in_sizes; (void)n_in; (void)out_size; (void)ws_size;
  const int B = 2, S = 2048, Dm = 1024, H = 16, OUT = 1024;

  const float* q = (const float*)d_in[0];
  const float* k = (const float*)d_in[1];
  const float* v = (const float*)d_in[2];
  const unsigned char* pmask = (const unsigned char*)d_in[3];
  const float* Wq = (const float*)d_in[4];
  const float* Wk = (const float*)d_in[5];
  const float* Wv = (const float*)d_in[6];
  const float* Wo = (const float*)d_in[7];
  float* out = (float*)d_out;

  // workspace layout (f16): 4 weight tensors (1M each) + Q/K/V/Ctx (4M each)
  const size_t WSZ = (size_t)1024 * 1024;
  const size_t QSZ = (size_t)B * S * 1024;
  _Float16* WqH = (_Float16*)d_ws;
  _Float16* WkH = WqH + WSZ;
  _Float16* WvH = WkH + WSZ;
  _Float16* WoH = WvH + WSZ;
  _Float16* Qb = WoH + WSZ;
  _Float16* Kb = Qb + QSZ;
  _Float16* Vb = Kb + QSZ;
  _Float16* Cx = Vb + QSZ;

  // 1) weights fp32 -> f16 (layouts are already B^T[N][K])
  cvt_f32_f16<<<dim3(1024, 4), 256, 0, stream>>>(Wq, Wk, Wv, Wo, WqH, WkH, WvH,
                                                 WoH, (int)WSZ);

  // 2) projections (softmax scale folded into Q)
  const int M = B * S; // 4096
  dim3 ggrid(1024 / BN, M / BM);
  gemm_f32a_f16c<<<ggrid, 256, 0, stream>>>(q, WqH, Qb, M, 1024, Dm, 0.125f);
  gemm_f32a_f16c<<<ggrid, 256, 0, stream>>>(k, WkH, Kb, M, 1024, Dm, 1.0f);
  gemm_f32a_f16c<<<ggrid, 256, 0, stream>>>(v, WvH, Vb, M, 1024, Dm, 1.0f);

  // 3) flash attention
  attn_kernel<<<dim3(S / ABR, H, B), 256, 0, stream>>>(Qb, Kb, Vb, pmask, Cx, S);

  // 4) output projection -> fp32 d_out
  gemm_f16a_f32c<<<ggrid, 256, 0, stream>>>(Cx, WoH, out, M, OUT, H * 64);
}
